// DependencyParser_53626961658022
// MI455X (gfx1250) — compile-verified, matrix operand order checked by reading, not run on hardware
//
#include <hip/hip_runtime.h>
#include <hip/hip_bf16.h>

// ---------------------------------------------------------------------------
// DependencyParser: embed -> 2-layer BiLSTM (H=400) -> pairwise tanh scores
// MI455X (gfx1250, wave32):
//   * input projections: v_wmma_f32_16x16x32_bf16, 16x64 tile per wave
//     (A-operand reused across 4 accumulators -> 4 WMMAs per A fetch)
//   * recurrent scans: 1 workgroup (32 waves) per direction, f32 VALU matvec
//     over L2-hot weights, global_prefetch_b8 on next-step gate row
// ---------------------------------------------------------------------------

typedef __attribute__((ext_vector_type(16))) __bf16 v16bf;
typedef __attribute__((ext_vector_type(8)))  float  v8f;

#define LSEQ 512
#define HN   400
#define H4   1600
#define KP0  416   // 400 padded to 13*32 for bf16 WMMA K-loop
#define KP1  800   // 25*32, no padding needed

static __device__ __forceinline__ float sigmoidf_(float x) {
  return 1.0f / (1.0f + __expf(-x));
}

// ---------------- weight / activation conversion --------------------------

// src: [rows, Kin] f32 row-major  ->  dst: [rows, Kp] bf16, zero-padded cols
__global__ void cvt_pad_bf16_kernel(const float* __restrict__ src,
                                    __bf16* __restrict__ dst,
                                    int rows, int Kin, int Kp) {
  int idx = blockIdx.x * blockDim.x + threadIdx.x;
  int total = rows * Kp;
  if (idx >= total) return;
  int r = idx / Kp;
  int c = idx - r * Kp;
  dst[idx] = (c < Kin) ? (__bf16)src[r * Kin + c] : (__bf16)0.0f;
}

__global__ void add_bias_kernel(const float* __restrict__ a,
                                const float* __restrict__ b,
                                float* __restrict__ o, int n) {
  int i = blockIdx.x * blockDim.x + threadIdx.x;
  if (i < n) o[i] = a[i] + b[i];
}

// Embedding gather + concat -> X0 bf16 [LSEQ, KP0] (cols 400..415 zeroed)
__global__ void embed_kernel(const int* __restrict__ widx,
                             const int* __restrict__ tidx,
                             const float* __restrict__ wemb,
                             const float* __restrict__ temb,
                             __bf16* __restrict__ X0) {
  int row = blockIdx.x;          // 0..511
  int col = threadIdx.x;         // 0..415 (13 waves)
  float v;
  if (col < 300)      v = wemb[(long)widx[row] * 300 + col];
  else if (col < 400) v = temb[(long)tidx[row] * 100 + (col - 300)];
  else                v = 0.0f;
  X0[row * KP0 + col] = (__bf16)v;
}

// ---------------- WMMA GEMM: C[M,N] = A[M,Kp] * W[N,Kp]^T + bias[N] --------
// One wave per 16x64 output tile (4 accumulators share one A fetch).
// blockIdx.y selects the LSTM direction. N must be a multiple of 64.
__global__ void wmma_gemm_bias_kernel(const __bf16* __restrict__ A,
                                      const __bf16* __restrict__ Wmat,
                                      const float*  __restrict__ bias,
                                      float* __restrict__ C,
                                      int M, int N, int Kp,
                                      long wStrideDir, long cStrideDir,
                                      int bStrideDir) {
  const int dir = blockIdx.y;
  const __bf16* W = Wmat + (long)dir * wStrideDir;
  const float*  B = bias + (long)dir * bStrideDir;
  float*        D = C    + (long)dir * cStrideDir;

  const int wavesPerBlock = blockDim.x >> 5;
  const int wid    = blockIdx.x * wavesPerBlock + (threadIdx.x >> 5);
  const int ngrp   = N >> 6;                 // groups of 64 columns
  const int total  = (M >> 4) * ngrp;
  if (wid >= total) return;                  // wave-uniform: EXEC stays all-1s

  const int tm  = wid / ngrp;
  const int tng = wid - tm * ngrp;
  const int lane  = threadIdx.x & 31;
  const int sub   = lane & 15;               // row-within-tile (A) / col (B)
  const int khalf = (lane >> 4) * 16;        // 16-wide K half this lane holds

  const __bf16* arow  = A + (long)(tm * 16 + sub) * Kp + khalf;
  const __bf16* wrow0 = W + (long)(tng * 64 +  0 + sub) * Kp + khalf;
  const __bf16* wrow1 = W + (long)(tng * 64 + 16 + sub) * Kp + khalf;
  const __bf16* wrow2 = W + (long)(tng * 64 + 32 + sub) * Kp + khalf;
  const __bf16* wrow3 = W + (long)(tng * 64 + 48 + sub) * Kp + khalf;

  v8f acc0 = {}, acc1 = {}, acc2 = {}, acc3 = {};
  for (int k = 0; k < Kp; k += 32) {
    v16bf a  = *(const v16bf*)(arow  + k);
    v16bf b0 = *(const v16bf*)(wrow0 + k);
    v16bf b1 = *(const v16bf*)(wrow1 + k);
    v16bf b2 = *(const v16bf*)(wrow2 + k);
    v16bf b3 = *(const v16bf*)(wrow3 + k);
    acc0 = __builtin_amdgcn_wmma_f32_16x16x32_bf16(false, a, false, b0,
                                                   (short)0, acc0, false, false);
    acc1 = __builtin_amdgcn_wmma_f32_16x16x32_bf16(false, a, false, b1,
                                                   (short)0, acc1, false, false);
    acc2 = __builtin_amdgcn_wmma_f32_16x16x32_bf16(false, a, false, b2,
                                                   (short)0, acc2, false, false);
    acc3 = __builtin_amdgcn_wmma_f32_16x16x32_bf16(false, a, false, b3,
                                                   (short)0, acc3, false, false);
  }

  // C/D layout: VGPR r, lane l -> M = r + 8*(l>>4), N = l&15
  const int col   = lane & 15;
  const int rbase = (lane >> 4) * 8;
  const v8f accs[4] = {acc0, acc1, acc2, acc3};
  #pragma unroll
  for (int j = 0; j < 4; ++j) {
    const int colbase = tng * 64 + j * 16;
    const float bcol = B[colbase + col];
    #pragma unroll
    for (int r = 0; r < 8; ++r) {
      int row = rbase + r;
      D[(long)(tm * 16 + row) * N + colbase + col] = accs[j][r] + bcol;
    }
  }
}

// ---------------- sequential LSTM scan (one workgroup per direction) ------
// G:   [2][LSEQ][H4] precomputed x@Wih^T + b_ih + b_hh
// Whh: [2][H4][HN] f32 (original input tensor)
// Hout:[LSEQ][2*HN] f32 concat (fwd | bwd);  Hbf optional bf16 mirror
// blockDim.x == 1024 (32 waves -> full WGP occupancy)
__global__ void lstm_scan_kernel(const float* __restrict__ G,
                                 const float* __restrict__ Whh,
                                 float* __restrict__ Hout,
                                 __bf16* __restrict__ Hbf) {
  const int dir = blockIdx.x;          // 0 = fwd, 1 = bwd
  const int tid = threadIdx.x;
  const bool rev = (dir == 1);
  const float* Gd = G   + (long)dir * LSEQ * H4;
  const float* Wd = Whh + (long)dir * H4 * HN;

  __shared__ float hsh[HN];
  __shared__ float csh[HN];
  __shared__ float gsh[H4];

  if (tid < HN) { hsh[tid] = 0.0f; csh[tid] = 0.0f; }
  __syncthreads();

  for (int t = 0; t < LSEQ; ++t) {
    const int tt  = rev ? (LSEQ - 1 - t) : t;
    const int ttn = rev ? (tt - 1) : (tt + 1);
    const float* gin = Gd + (long)tt * H4;
    const float* ginNext = Gd + (long)ttn * H4;
    const bool haveNext = (t + 1 < LSEQ);

    // g = G[tt] + h @ Whh^T   (1024 threads; rows r and r+1024)
    for (int r = tid; r < H4; r += 1024) {
      const float4* w4 = (const float4*)(Wd + (long)r * HN);
      if (haveNext) __builtin_prefetch(ginNext + r, 0, 0);  // next-step gates
      float sum = gin[r];
      #pragma unroll 4
      for (int k = 0; k < HN / 4; ++k) {
        float4 wv = w4[k];
        sum += hsh[4 * k + 0] * wv.x + hsh[4 * k + 1] * wv.y +
               hsh[4 * k + 2] * wv.z + hsh[4 * k + 3] * wv.w;
      }
      gsh[r] = sum;
    }
    __syncthreads();

    if (tid < HN) {
      float ig = sigmoidf_(gsh[tid]);
      float fg = sigmoidf_(gsh[HN + tid]);
      float gg = tanhf(gsh[2 * HN + tid]);
      float og = sigmoidf_(gsh[3 * HN + tid]);
      float cn = fg * csh[tid] + ig * gg;
      float hn = og * tanhf(cn);
      csh[tid] = cn;
      hsh[tid] = hn;
      Hout[(long)tt * (2 * HN) + dir * HN + tid] = hn;
      if (Hbf) Hbf[(long)tt * (2 * HN) + dir * HN + tid] = (__bf16)hn;
    }
    __syncthreads();
  }
}

// ---------------- epilogue -------------------------------------------------

// st[0..511] = H1 @ fc1_w[:, :800]^T ; st[512..1023] = H1 @ fc1_w[:, 800:]^T
__global__ void head_dep_kernel(const float* __restrict__ H1,
                                const float* __restrict__ fcw,
                                float* __restrict__ st) {
  int t = blockIdx.x * blockDim.x + threadIdx.x;
  if (t >= LSEQ) return;
  const float* hr = H1 + (long)t * (2 * HN);
  float s = 0.0f, d = 0.0f;
  #pragma unroll 4
  for (int k = 0; k < 2 * HN; ++k) {
    float hv = hr[k];
    s += hv * fcw[k];
    d += hv * fcw[2 * HN + k];
  }
  st[t] = s;
  st[LSEQ + t] = d;
}

__global__ void scores_kernel(const float* __restrict__ st,
                              const float* __restrict__ fcb,
                              float* __restrict__ out) {
  int idx = blockIdx.x * blockDim.x + threadIdx.x;
  if (idx >= LSEQ * LSEQ) return;
  int i = idx >> 9;       // / 512
  int j = idx & (LSEQ - 1);
  out[idx] = tanhf(st[i] + st[LSEQ + j] + fcb[0]);
}

// ---------------------------------------------------------------------------

static inline size_t align_up256(size_t x) { return (x + 255) & ~(size_t)255; }

extern "C" void kernel_launch(void* const* d_in, const int* in_sizes, int n_in,
                              void* d_out, int out_size, void* d_ws, size_t ws_size,
                              hipStream_t stream) {
  (void)in_sizes; (void)n_in; (void)out_size; (void)ws_size;

  const int*   widx   = (const int*)  d_in[0];
  const int*   tidx   = (const int*)  d_in[1];
  const float* wemb   = (const float*)d_in[3];
  const float* temb   = (const float*)d_in[4];
  const float* wih0   = (const float*)d_in[5];   // [2,1600,400]
  const float* whh0   = (const float*)d_in[6];   // [2,1600,400]
  const float* bih0   = (const float*)d_in[7];
  const float* bhh0   = (const float*)d_in[8];
  const float* wih1   = (const float*)d_in[9];   // [2,1600,800]
  const float* whh1   = (const float*)d_in[10];  // [2,1600,400]
  const float* bih1   = (const float*)d_in[11];
  const float* bhh1   = (const float*)d_in[12];
  const float* fc1w   = (const float*)d_in[13];  // [1,1600]
  const float* fc1b   = (const float*)d_in[14];  // [1]
  float* out = (float*)d_out;

  // ---- workspace carve-up (all 256B aligned) ----
  char* ws = (char*)d_ws;
  size_t off = 0;
  __bf16* X0   = (__bf16*)(ws + off); off = align_up256(off + (size_t)LSEQ * KP0 * 2);
  __bf16* WIH0 = (__bf16*)(ws + off); off = align_up256(off + (size_t)2 * H4 * KP0 * 2);
  __bf16* WIH1 = (__bf16*)(ws + off); off = align_up256(off + (size_t)2 * H4 * KP1 * 2);
  float*  B0   = (float*) (ws + off); off = align_up256(off + (size_t)2 * H4 * 4);
  float*  B1   = (float*) (ws + off); off = align_up256(off + (size_t)2 * H4 * 4);
  float*  G0   = (float*) (ws + off); off = align_up256(off + (size_t)2 * LSEQ * H4 * 4);
  float*  G1   = (float*) (ws + off); off = align_up256(off + (size_t)2 * LSEQ * H4 * 4);
  float*  H0f  = (float*) (ws + off); off = align_up256(off + (size_t)LSEQ * 2 * HN * 4);
  __bf16* H0bf = (__bf16*)(ws + off); off = align_up256(off + (size_t)LSEQ * 2 * HN * 2);
  float*  H1f  = (float*) (ws + off); off = align_up256(off + (size_t)LSEQ * 2 * HN * 4);
  float*  ST   = (float*) (ws + off); off = align_up256(off + (size_t)2 * LSEQ * 4);

  // ---- weight prep ----
  {
    int tot0 = 2 * H4 * KP0;
    cvt_pad_bf16_kernel<<<(tot0 + 255) / 256, 256, 0, stream>>>(wih0, WIH0, 2 * H4, 400, KP0);
    int tot1 = 2 * H4 * KP1;
    cvt_pad_bf16_kernel<<<(tot1 + 255) / 256, 256, 0, stream>>>(wih1, WIH1, 2 * H4, 800, KP1);
    add_bias_kernel<<<(2 * H4 + 255) / 256, 256, 0, stream>>>(bih0, bhh0, B0, 2 * H4);
    add_bias_kernel<<<(2 * H4 + 255) / 256, 256, 0, stream>>>(bih1, bhh1, B1, 2 * H4);
  }

  // ---- embeddings -> X0 bf16 [512,416] ----
  embed_kernel<<<LSEQ, KP0, 0, stream>>>(widx, tidx, wemb, temb, X0);

  // ---- layer 0: input projection GEMM (WMMA 16x64/wave), then scan ----
  {
    int waveJobs = (LSEQ / 16) * (H4 / 64);          // 32 * 25 = 800 per dir
    dim3 grid((waveJobs + 7) / 8, 2);
    wmma_gemm_bias_kernel<<<grid, 256, 0, stream>>>(
        X0, WIH0, B0, G0, LSEQ, H4, KP0,
        (long)H4 * KP0, (long)LSEQ * H4, H4);
  }
  lstm_scan_kernel<<<2, 1024, 0, stream>>>(G0, whh0, H0f, H0bf);

  // ---- layer 1: input projection GEMM (WMMA 16x64/wave), then scan ----
  {
    int waveJobs = (LSEQ / 16) * (H4 / 64);
    dim3 grid((waveJobs + 7) / 8, 2);
    wmma_gemm_bias_kernel<<<grid, 256, 0, stream>>>(
        H0bf, WIH1, B1, G1, LSEQ, H4, KP1,
        (long)H4 * KP1, (long)LSEQ * H4, H4);
  }
  lstm_scan_kernel<<<2, 1024, 0, stream>>>(G1, whh1, H1f, (__bf16*)nullptr);

  // ---- epilogue: head/dep projections + pairwise tanh scores ----
  head_dep_kernel<<<(LSEQ + 255) / 256, 256, 0, stream>>>(H1f, fc1w, ST);
  scores_kernel<<<(LSEQ * LSEQ + 255) / 256, 256, 0, stream>>>(ST, fc1b, out);
  (void)H0f; (void)off;
}
